// HRANEncoder_30236569763851
// MI455X (gfx1250) — compile-verified, matrix operand order checked by reading, not run on hardware
//
#include <hip/hip_runtime.h>
#include <stdint.h>

#define DD 256
#define GG 768            // 3*D
#define NB 64
#define TT 512
#define NROWS (NB * TT)   // 32768
#define NTB 4             // N-tiles per wave (register blocking)

typedef __attribute__((ext_vector_type(16))) __bf16 v16bf;
typedef __attribute__((ext_vector_type(8)))  float  v8f;

__device__ __forceinline__ float bf2f(uint32_t lo16) {
    return __uint_as_float(lo16 << 16);
}

// ---------------------------------------------------------------------------
// Prep: convert wg_Whh (768x256 fp32) -> bf16 (RNE via v_cvt_pk_bf16_f32)
// ---------------------------------------------------------------------------
__global__ void __launch_bounds__(256) conv_bf16_kernel(
    const float* __restrict__ src, uint16_t* __restrict__ dst, int n) {
    int i = blockIdx.x * blockDim.x + threadIdx.x;
    if (i < n) {
        __bf16 v = (__bf16)src[i];
        dst[i] = __builtin_bit_cast(uint16_t, v);
    }
}

// ---------------------------------------------------------------------------
// Phase A: gx[row,g] = sum_d embed[tokens[row],d] * W_ih[g,d] + b_ih[g]
// 32768x768x256 GEMM via v_wmma_f32_16x16x32_bf16, fused embedding gather.
// One wave computes a 16x64 output strip (4 N-tiles): A fragments (the
// gathered embedding rows) are loaded once and reused 4x -> 32 WMMAs/wave.
// ---------------------------------------------------------------------------
__global__ void __launch_bounds__(256) gx_wmma_kernel(
    const int*   __restrict__ tokens, const float* __restrict__ embed,
    const float* __restrict__ Wih,    const float* __restrict__ bih,
    float* __restrict__ gx)
{
    const int lane  = threadIdx.x & 31;
    const int wave  = threadIdx.x >> 5;
    const int strip = blockIdx.x * 8 + wave;
    const int NGRP  = GG / (16 * NTB);     // 12 N-groups of 64 cols
    const int ngrp  = strip % NGRP;
    const int mtile = strip / NGRP;
    const int l16   = lane & 15;
    const int hi    = lane >> 4;           // lane-group selects K half

    const int row = mtile * 16 + l16;      // A-matrix row supplied by this lane
    const int tok = tokens[row];
    const float* arow = embed + (size_t)tok * DD;
    const float* brow[NTB];
    #pragma unroll
    for (int nt = 0; nt < NTB; ++nt)       // B col = W_ih row
        brow[nt] = Wih + (size_t)(ngrp * 64 + nt * 16 + l16) * DD;

    v8f acc[NTB] = {};
    #pragma unroll
    for (int k0 = 0; k0 < DD; k0 += 32) {
        v16bf a;
        // 16-bit A/B fragment layout: elems 0..7 = K{k0+hi*8 ..}, 8..15 = +16
        #pragma unroll
        for (int i = 0; i < 8; ++i) {
            a[i]     = (__bf16)arow[k0 + hi * 8 + i];
            a[i + 8] = (__bf16)arow[k0 + 16 + hi * 8 + i];
        }
        #pragma unroll
        for (int nt = 0; nt < NTB; ++nt) {
            v16bf b;
            #pragma unroll
            for (int i = 0; i < 8; ++i) {
                b[i]     = (__bf16)brow[nt][k0 + hi * 8 + i];
                b[i + 8] = (__bf16)brow[nt][k0 + 16 + hi * 8 + i];
            }
            acc[nt] = __builtin_amdgcn_wmma_f32_16x16x32_bf16(
                false, a, false, b, (short)0, acc[nt], false, false);
        }
    }

    #pragma unroll
    for (int nt = 0; nt < NTB; ++nt) {
        const int   col  = ngrp * 64 + nt * 16 + l16; // C/D: N=lane&15, M=v+hi*8
        const float bias = bih[col];
        #pragma unroll
        for (int v = 0; v < 8; ++v) {
            int orow = mtile * 16 + hi * 8 + v;
            gx[(size_t)orow * GG + col] = acc[nt][v] + bias;
        }
    }
}

// ---------------------------------------------------------------------------
// Phase B: per-utterance GRU scan + fused online-softmax attention pool.
// One block (256 thr) per utterance; thread j owns hidden dim j.
// W_hh streamed from L2 in bf16 (384KB/step/WG).
// ---------------------------------------------------------------------------
__global__ void __launch_bounds__(256) word_gru_kernel(
    const float* __restrict__ gx, const uint16_t* __restrict__ whh,
    const float* __restrict__ bhh, const float* __restrict__ uaw_p,
    float* __restrict__ utt)
{
    __shared__ float hsm[DD];
    __shared__ float red[8];
    const int b    = blockIdx.x;
    const int j    = threadIdx.x;
    const int lane = j & 31;
    const int wv   = j >> 5;

    const float bhr = bhh[j], bhz = bhh[DD + j], bhn = bhh[2 * DD + j];
    const float uaw = uaw_p[j];
    const uint4* wr = (const uint4*)(whh + (size_t)j * DD);
    const uint4* wz = (const uint4*)(whh + (size_t)(DD + j) * DD);
    const uint4* wn = (const uint4*)(whh + (size_t)(2 * DD + j) * DD);

    hsm[j] = 0.0f;
    float m = -3.0e38f, s = 0.0f, acc = 0.0f;
    __syncthreads();

    const float* gbase = gx + (size_t)b * TT * GG;
    for (int t = 0; t < TT; ++t) {
        const float* gp = gbase + (size_t)t * GG;
        if (t + 1 < TT) __builtin_prefetch(&gp[GG + j], 0, 1); // next-step gx

        float dr = 0.f, dz = 0.f, dn = 0.f;
        #pragma unroll 4
        for (int q = 0; q < DD / 8; ++q) {
            uint4 a = wr[q], c = wz[q], e = wn[q];
            const float* hp = hsm + q * 8;
            float h0 = hp[0], h1 = hp[1], h2 = hp[2], h3 = hp[3];
            float h4 = hp[4], h5 = hp[5], h6 = hp[6], h7 = hp[7];
            dr += bf2f(a.x & 0xffffu) * h0 + bf2f(a.x >> 16) * h1
                + bf2f(a.y & 0xffffu) * h2 + bf2f(a.y >> 16) * h3
                + bf2f(a.z & 0xffffu) * h4 + bf2f(a.z >> 16) * h5
                + bf2f(a.w & 0xffffu) * h6 + bf2f(a.w >> 16) * h7;
            dz += bf2f(c.x & 0xffffu) * h0 + bf2f(c.x >> 16) * h1
                + bf2f(c.y & 0xffffu) * h2 + bf2f(c.y >> 16) * h3
                + bf2f(c.z & 0xffffu) * h4 + bf2f(c.z >> 16) * h5
                + bf2f(c.w & 0xffffu) * h6 + bf2f(c.w >> 16) * h7;
            dn += bf2f(e.x & 0xffffu) * h0 + bf2f(e.x >> 16) * h1
                + bf2f(e.y & 0xffffu) * h2 + bf2f(e.y >> 16) * h3
                + bf2f(e.z & 0xffffu) * h4 + bf2f(e.z >> 16) * h5
                + bf2f(e.w & 0xffffu) * h6 + bf2f(e.w >> 16) * h7;
        }

        float xr = gp[j], xz = gp[DD + j], xn = gp[2 * DD + j];
        float r = 1.f / (1.f + __expf(-(xr + dr + bhr)));
        float z = 1.f / (1.f + __expf(-(xz + dz + bhz)));
        float n = tanhf(xn + r * (dn + bhn));
        float hnew = (1.f - z) * n + z * hsm[j];

        // logit l_t = dot(h_t, ua_w): wave32 shuffle tree + cross-wave LDS
        float p = hnew * uaw;
        #pragma unroll
        for (int o = 16; o > 0; o >>= 1) p += __shfl_xor(p, o, 32);
        __syncthreads();                 // all hsm reads of this step done
        hsm[j] = hnew;
        if (lane == 0) red[wv] = p;
        __syncthreads();
        float l = red[0] + red[1] + red[2] + red[3]
                + red[4] + red[5] + red[6] + red[7];

        // online softmax accumulation of sum_t softmax(l)_t * h_t
        float mn   = fmaxf(m, l);
        float corr = __expf(m - mn);
        float w    = __expf(l - mn);
        s   = s * corr + w;
        acc = acc * corr + w * hnew;
        m   = mn;
    }
    utt[b * DD + j] = acc / s;
}

// ---------------------------------------------------------------------------
// Phase C: sentence GRU with T=1, h0=0  =>  out = (1-z)*n ; attn_pool(T=1)=id
// ---------------------------------------------------------------------------
__global__ void __launch_bounds__(256) sent_final_kernel(
    const float* __restrict__ utt, const float* __restrict__ Wih,
    const float* __restrict__ bih, const float* __restrict__ bhh,
    float* __restrict__ out)
{
    __shared__ float uv[DD];
    const int b = blockIdx.x, j = threadIdx.x;
    uv[j] = utt[b * DD + j];
    __syncthreads();

    float xr = bih[j], xz = bih[DD + j], xn = bih[2 * DD + j];
    const float* wrr = Wih + (size_t)j * DD;
    const float* wzz = Wih + (size_t)(DD + j) * DD;
    const float* wnn = Wih + (size_t)(2 * DD + j) * DD;
    for (int d = 0; d < DD; ++d) {
        float u = uv[d];
        xr += wrr[d] * u; xz += wzz[d] * u; xn += wnn[d] * u;
    }
    float r = 1.f / (1.f + __expf(-(xr + bhh[j])));
    float z = 1.f / (1.f + __expf(-(xz + bhh[DD + j])));
    float n = tanhf(xn + r * bhh[2 * DD + j]);
    out[b * DD + j] = (1.f - z) * n;
}

// ---------------------------------------------------------------------------
extern "C" void kernel_launch(void* const* d_in, const int* in_sizes, int n_in,
                              void* d_out, int out_size, void* d_ws, size_t ws_size,
                              hipStream_t stream) {
    const int*   tokens = (const int*)  d_in[0];
    const float* embed  = (const float*)d_in[1];
    const float* wgWih  = (const float*)d_in[2];
    const float* wgWhh  = (const float*)d_in[3];
    const float* wgbih  = (const float*)d_in[4];
    const float* wgbhh  = (const float*)d_in[5];
    const float* uaw    = (const float*)d_in[6];
    // d_in[7] = ua_b   : cancels in softmax
    const float* sgWih  = (const float*)d_in[8];
    // d_in[9] = sg_Whh : multiplied by h0 == 0
    const float* sgbih  = (const float*)d_in[10];
    const float* sgbhh  = (const float*)d_in[11];
    // d_in[12..13] = da_w, da_b : attn over T=1 is identity
    (void)in_sizes; (void)n_in; (void)out_size; (void)ws_size;

    char* ws = (char*)d_ws;
    float*    gx   = (float*)ws;                               // 96 MB
    size_t    off  = (size_t)NROWS * GG * sizeof(float);
    uint16_t* whhb = (uint16_t*)(ws + off);                    // 384 KB
    off += (size_t)GG * DD * sizeof(uint16_t);
    float*    utt  = (float*)(ws + off);                       // 64 KB
    float*    out  = (float*)d_out;

    conv_bf16_kernel<<<(GG * DD) / 256, 256, 0, stream>>>(wgWhh, whhb, GG * DD);

    const int strips = (NROWS / 16) * (GG / (16 * NTB));       // 2048*12 = 24576
    gx_wmma_kernel<<<strips / 8, 256, 0, stream>>>(tokens, embed, wgWih, wgbih, gx);

    word_gru_kernel<<<NB, DD, 0, stream>>>(gx, whhb, wgbhh, uaw, utt);

    sent_final_kernel<<<NB, DD, 0, stream>>>(utt, sgWih, sgbih, sgbhh, out);
}
